// EmotionEncoder_3959959847558
// MI455X (gfx1250) — matrix operand order repaired and checked by reading, not compiled, not run on hardware
//
#include <hip/hip_runtime.h>
#include <hip/hip_bf16.h>
#include <math.h>

typedef __attribute__((ext_vector_type(16))) __bf16 v16bf;
typedef __attribute__((ext_vector_type(8)))  __bf16 v8bf;
typedef __attribute__((ext_vector_type(8)))  float  v8f;

#define HID 768
#define HEADD 48
#define S3 3
#define NB 16384
#define ROWS (NB*S3)
#define LAMBDA_INIT 0.2f      // 0.8 - 0.6*exp(-0.3*0)
#define EPSV 1e-5f

// A-matrix (16x32, 16-bit) K offset for VGPR j, lane-half (ISA 7.12.2):
// lanes 0-15 hold K {0..7, 16..23}, lanes 16-31 hold K {8..15, 24..31}.
__device__ __forceinline__ int a_koff(int j, int half) {
  return ((j < 4) ? (2*j) : (16 + 2*(j-4))) + half*8;
}

// Build a 16x32 bf16 A fragment from a row-major LDS row: per lane this is
// exactly two contiguous 16-byte runs -> two ds_load_b128 + shuffle.
__device__ __forceinline__ v16bf load_a_frag(const __bf16* rowp, int kbase, int half) {
  v8bf lo = *(const v8bf*)(rowp + kbase + half*8);        // K = kbase+half*8 .. +7
  v8bf hi = *(const v8bf*)(rowp + kbase + 16 + half*8);   // K = kbase+16+half*8 .. +7
  return __builtin_shufflevector(lo, hi, 0,1,2,3,4,5,6,7,8,9,10,11,12,13,14,15);
}

__device__ __forceinline__ v8f wmma_bf16(v16bf a, v16bf b, v8f c) {
  return __builtin_amdgcn_wmma_f32_16x16x32_bf16(false, a, false, b, (short)0, c, false, false);
}

// ---------------------------------------------------------------------------
// Pack a row-major f32 weight (Kdim x 768) into bf16 WMMA-B fragment order:
// dst[((nt*KT + kk)*32 + lane)*16 + e] = W[kk*32 + (lane>>4)*8 + (e&7) + (e>>3)*16][nt*16 + (lane&15)]
// so each lane loads one aligned, contiguous v16bf per WMMA step.
// ---------------------------------------------------------------------------
__global__ __launch_bounds__(256)
void pack_b_kernel(const float* __restrict__ src, __bf16* __restrict__ dst, int Kdim) {
  int KT = Kdim >> 5;
  int total = 48 * KT * 512;
  int i = blockIdx.x*blockDim.x + threadIdx.x;
  if (i >= total) return;
  int e    = i & 15;
  int lane = (i >> 4) & 31;
  int tile = i >> 9;               // nt*KT + kk
  int nt = tile / KT, kk = tile % KT;
  int k = kk*32 + (lane >> 4)*8 + (e & 7) + (e >> 3)*16;
  int n = nt*16 + (lane & 15);
  dst[i] = (__bf16)src[(long)k*HID + n];
}

// ---------------------------------------------------------------------------
// Kernel 1: h = GELU(LN(emb @ W1 + b1)), written as bf16.
// One block = 16 rows; K=32 -> single WMMA per N-tile; 48 N-tiles / 8 waves.
// ---------------------------------------------------------------------------
__global__ __launch_bounds__(256)
void k_embed_ln_gelu(const int* __restrict__ eidx, const float* __restrict__ conf,
                     const float* __restrict__ etab, const __bf16* __restrict__ W1p,
                     const float* __restrict__ b1, const float* __restrict__ g,
                     const float* __restrict__ beta, __bf16* __restrict__ hbuf)
{
  __shared__ float sh[16*HID];
  __shared__ float red_s[16][16];
  __shared__ float red_q[16][16];
  __shared__ float s_mu[16], s_rs[16];

  const int t = threadIdx.x;
  const int lane = t & 31, wave = t >> 5;
  const int half = lane >> 4, m = lane & 15;
  const int rowbase = blockIdx.x * 16;

  // A fragment: gathered embedding * confidence (per-wave redundant; tiny).
  int row = rowbase + m;
  int id = eidx[row]; if (id < 0) id = 8;           // pad index -> zero row
  float cf = conf[row];
  v16bf a;
  #pragma unroll
  for (int j = 0; j < 8; ++j) {
    int k = a_koff(j, half);
    a[2*j]   = (__bf16)(etab[id*32 + k]     * cf);
    a[2*j+1] = (__bf16)(etab[id*32 + k + 1] * cf);
  }

  const v16bf* Wp = (const v16bf*)W1p;   // [nt][lane] fragments (KT==1)
  for (int tt = 0; tt < 6; ++tt) {
    int nt = wave*6 + tt;
    int col = nt*16 + m;
    v16bf b = Wp[nt*32 + lane];
    v8f c = {};
    c = wmma_bf16(a, b, c);
    #pragma unroll
    for (int i = 0; i < 8; ++i)
      sh[(i + (half ? 8 : 0))*HID + col] = c[i];   // C/D layout: M split at lane 16
  }
  __syncthreads();

  // LayerNorm (+b1) then exact GELU. 16 threads per row, 48 cols each.
  int r = t >> 4, cseg = t & 15;
  float s0 = 0.f, s1 = 0.f;
  for (int cc = cseg*48; cc < cseg*48+48; ++cc) {
    float x = sh[r*HID+cc] + b1[cc];
    s0 += x; s1 += x*x;
  }
  red_s[r][cseg] = s0; red_q[r][cseg] = s1;
  __syncthreads();
  if (cseg == 0) {
    float a0 = 0.f, a1 = 0.f;
    for (int i = 0; i < 16; ++i) { a0 += red_s[r][i]; a1 += red_q[r][i]; }
    float mu = a0 / (float)HID;
    float var = a1 / (float)HID - mu*mu;
    s_mu[r] = mu; s_rs[r] = rsqrtf(var + EPSV);
  }
  __syncthreads();
  float mu = s_mu[r], rs = s_rs[r];
  long grow = (long)(rowbase + r) * HID;
  for (int cc = cseg*48; cc < cseg*48+48; ++cc) {
    float x = sh[r*HID+cc] + b1[cc];
    float y = (x - mu) * rs * g[cc] + beta[cc];
    float z = 0.5f * y * (1.0f + erff(y * 0.70710678118f));
    hbuf[grow + cc] = (__bf16)z;
  }
}

// ---------------------------------------------------------------------------
// Kernel 2: fused QKV GEMMs + RoPE + differential attention + RMS-norm +
// Wo GEMM + masked mean.  One block = 4 batch items (12 rows, padded to 16).
// LDS overlaid by lifetime: raw_h holds h then o; raw_qkv holds q/k/v (bf16)
// then the f32 Wo-output tile.
// ---------------------------------------------------------------------------
__global__ __launch_bounds__(256)
void k_attn_out(const __bf16* __restrict__ hbuf,
                const __bf16* __restrict__ Wqp, const __bf16* __restrict__ Wkp,
                const __bf16* __restrict__ Wvp, const __bf16* __restrict__ Wop,
                const int* __restrict__ eidx,
                const float* __restrict__ lq1, const float* __restrict__ lk1,
                const float* __restrict__ lq2, const float* __restrict__ lk2,
                const float* __restrict__ subw,
                float* __restrict__ out)
{
  __shared__ __align__(16) unsigned char raw_h[16*HID*2];       // 24.0 KB
  __shared__ __align__(16) unsigned char raw_qkv[3*16*HID*2];   // 72.0 KB
  __shared__ float s_cos[S3][24], s_sin[S3][24];
  __shared__ float s_lam;
  __shared__ int   s_pad[12];
  __shared__ float s_den[4];

  __bf16* sh   = (__bf16*)raw_h;
  __bf16* so   = (__bf16*)raw_h;        // o tile reuses h space
  __bf16* sqkv = (__bf16*)raw_qkv;
  float*  sout = (float*)raw_qkv;       // f32 Wo-output tile reuses q/k space

  const int t = threadIdx.x;
  const int lane = t & 31, wave = t >> 5;
  const int half = lane >> 4, m = lane & 15;
  const int b0 = blockIdx.x * 4;
  const long rowbase = (long)b0 * 3;

  // Stage h tile (12 valid rows, rows 12..15 zero).
  for (int i = t; i < 16*HID; i += 256) {
    int mm = i / HID, cc = i % HID;
    sh[i] = (mm < 12) ? hbuf[(rowbase + mm)*HID + cc] : (__bf16)0.0f;
  }
  if (t < 72) {  // RoPE tables: 3 positions x 24 freq pairs
    int s = t / 24, j = t % 24;
    float f = expf(-(2.0f*(float)j/48.0f) * 9.2103403719762f);  // ln(10000)
    float ang = (float)s * f;
    s_cos[s][j] = cosf(ang); s_sin[s][j] = sinf(ang);
  }
  if (t < 12)
    s_pad[t] = (eidx[(b0 + t/3)*3 + (t%3)] < 0) ? 1 : 0;
  if (t == 0) {
    float d1 = 0.f, d2 = 0.f;
    for (int i = 0; i < HEADD; ++i) { d1 += lq1[i]*lk1[i]; d2 += lq2[i]*lk2[i]; }
    s_lam = expf(d1) - expf(d2) + LAMBDA_INIT;
  }
  __syncthreads();
  if (t < 4) {
    int c = 0;
    for (int s = 0; s < 3; ++s) c += 1 - s_pad[t*3 + s];
    s_den[t] = fmaxf((float)c, 1.0f);
  }

  // ---- Q/K/V GEMMs: 3 mats x 48 N-tiles = 144 tiles, processed in pairs
  //      sharing one A fragment; K=768 in 24 WMMA steps.
  for (int jp = wave; jp < 72; jp += 8) {
    int job0 = jp*2;
    int mat = job0 / 48;                   // pair never straddles a matrix
    const v16bf* Wp = (const v16bf*)((mat == 0) ? Wqp : (mat == 1) ? Wkp : Wvp);
    int nt0 = job0 % 48, nt1 = nt0 + 1;
    v8f c0 = {}, c1 = {};
    for (int kk = 0; kk < 24; ++kk) {
      v16bf a  = load_a_frag(sh + m*HID, kk*32, half);
      v16bf bf0 = Wp[((long)nt0*24 + kk)*32 + lane];
      v16bf bf1 = Wp[((long)nt1*24 + kk)*32 + lane];
      __builtin_prefetch(&Wp[((long)nt0*24 + kk + 1)*32 + lane], 0, 0);
      c0 = wmma_bf16(a, bf0, c0);
      c1 = wmma_bf16(a, bf1, c1);
    }
    __bf16* dst = sqkv + mat*16*HID;
    #pragma unroll
    for (int i = 0; i < 8; ++i) {
      int mm = i + (half ? 8 : 0);
      dst[mm*HID + nt0*16 + m] = (__bf16)c0[i];
      dst[mm*HID + nt1*16 + m] = (__bf16)c1[i];
    }
  }
  __syncthreads();

  // ---- RoPE on q and k, in place in LDS ----
  for (int i = t; i < 2*16*16*24; i += 256) {
    int mat = i / (16*16*24);
    int rem = i % (16*16*24);
    int mm  = rem / (16*24);
    int r2  = rem % (16*24);
    int hh  = r2 / 24, j = r2 % 24;
    int s   = mm % 3;
    __bf16* p = sqkv + mat*16*HID + mm*HID + hh*48 + 2*j;
    float x1 = (float)p[0], x2 = (float)p[1];
    float cs = s_cos[s][j], sn = s_sin[s][j];
    p[0] = (__bf16)(x1*cs - x2*sn);
    p[1] = (__bf16)(x1*sn + x2*cs);
  }
  __syncthreads();

  // ---- Differential attention + RMS-norm (3x3 scores per sub-head) ----
  const float lam = s_lam;
  const float scale = 0.14433756729740643f;  // 48^-0.5
  if (t < 96) {
    int bl = t / 24, rem = t % 24;
    int h8 = rem / 3, qs = rem % 3;
    int qrow = bl*3 + qs;
    const __bf16* q = sqkv + 0*16*HID;
    const __bf16* k = sqkv + 1*16*HID;
    const __bf16* v = sqkv + 2*16*HID;
    float attn[2][3];
    #pragma unroll
    for (int c = 0; c < 2; ++c) {
      int hh = h8*2 + c;
      float sc[3]; float mx = -1e30f;
      for (int ks = 0; ks < 3; ++ks) {
        int krow = bl*3 + ks;
        float d = 0.f;
        for (int dd = 0; dd < 48; ++dd)
          d += (float)q[qrow*HID + hh*48 + dd] * (float)k[krow*HID + hh*48 + dd];
        d = d*scale + (s_pad[krow] ? -1e30f : 0.f);
        sc[ks] = d; mx = fmaxf(mx, d);
      }
      float sum = 0.f;
      for (int ks = 0; ks < 3; ++ks) { sc[ks] = expf(sc[ks]-mx); sum += sc[ks]; }
      float inv = 1.0f / sum;
      for (int ks = 0; ks < 3; ++ks) attn[c][ks] = sc[ks]*inv;
    }
    float diff[3];
    for (int ks = 0; ks < 3; ++ks) diff[ks] = attn[0][ks] - lam*attn[1][ks];
    // two passes: RMS then scaled store (avoids a 96-float register array)
    float ssq = 0.f;
    for (int dd = 0; dd < 96; ++dd) {
      float acc = 0.f;
      for (int ks = 0; ks < 3; ++ks)
        acc += diff[ks] * (float)v[(bl*3+ks)*HID + h8*96 + dd];
      ssq += acc*acc;
    }
    float rinv = rsqrtf(ssq/96.f + EPSV) * (1.0f - LAMBDA_INIT);
    for (int dd = 0; dd < 96; ++dd) {
      float acc = 0.f;
      for (int ks = 0; ks < 3; ++ks)
        acc += diff[ks] * (float)v[(bl*3+ks)*HID + h8*96 + dd];
      so[qrow*HID + h8*96 + dd] = (__bf16)(acc * rinv * subw[dd]);
    }
  }
  __syncthreads();

  // ---- Wo GEMM: o(16x768) @ Wo(768x768) -> f32 tile in LDS, tile pairs ----
  for (int jp = wave; jp < 24; jp += 8) {
    int nt0 = jp*2, nt1 = nt0 + 1;
    const v16bf* Wp = (const v16bf*)Wop;
    v8f c0 = {}, c1 = {};
    for (int kk = 0; kk < 24; ++kk) {
      v16bf a  = load_a_frag(so + m*HID, kk*32, half);
      v16bf bf0 = Wp[((long)nt0*24 + kk)*32 + lane];
      v16bf bf1 = Wp[((long)nt1*24 + kk)*32 + lane];
      __builtin_prefetch(&Wp[((long)nt0*24 + kk + 1)*32 + lane], 0, 0);
      c0 = wmma_bf16(a, bf0, c0);
      c1 = wmma_bf16(a, bf1, c1);
    }
    #pragma unroll
    for (int i = 0; i < 8; ++i) {
      int mm = i + (half ? 8 : 0);
      sout[mm*HID + nt0*16 + m] = c0[i];
      sout[mm*HID + nt1*16 + m] = c1[i];
    }
  }
  __syncthreads();

  // ---- masked mean over S=3, write final f32 output ----
  for (int i = t; i < 4*HID; i += 256) {
    int bl = i / HID, n = i % HID;
    float acc = 0.f;
    for (int s = 0; s < 3; ++s)
      if (!s_pad[bl*3 + s]) acc += sout[(bl*3 + s)*HID + n];
    out[((long)(b0 + bl))*HID + n] = acc / s_den[bl];
  }
}

// ---------------------------------------------------------------------------
extern "C" void kernel_launch(void* const* d_in, const int* in_sizes, int n_in,
                              void* d_out, int out_size, void* d_ws, size_t ws_size,
                              hipStream_t stream) {
  (void)in_sizes; (void)n_in; (void)out_size; (void)ws_size;
  const int*   eidx = (const int*)  d_in[0];
  const float* conf = (const float*)d_in[1];
  const float* etab = (const float*)d_in[2];
  const float* W1   = (const float*)d_in[3];
  const float* b1   = (const float*)d_in[4];
  const float* lng  = (const float*)d_in[5];
  const float* lnb  = (const float*)d_in[6];
  const float* Wq   = (const float*)d_in[7];
  const float* Wk   = (const float*)d_in[8];
  const float* Wv   = (const float*)d_in[9];
  const float* Wo   = (const float*)d_in[10];
  const float* lq1  = (const float*)d_in[11];
  const float* lk1  = (const float*)d_in[12];
  const float* lq2  = (const float*)d_in[13];
  const float* lk2  = (const float*)d_in[14];
  const float* subw = (const float*)d_in[15];
  float* out = (float*)d_out;

  __bf16* w = (__bf16*)d_ws;
  __bf16* W1p = w;  w += 32*HID;
  __bf16* Wqp = w;  w += HID*HID;
  __bf16* Wkp = w;  w += HID*HID;
  __bf16* Wvp = w;  w += HID*HID;
  __bf16* Wop = w;  w += HID*HID;
  __bf16* hb  = w;  // ROWS*HID bf16 (~75.5 MB)

  pack_b_kernel<<<(32*HID + 255)/256, 256, 0, stream>>>(W1, W1p, 32);
  pack_b_kernel<<<(HID*HID + 255)/256, 256, 0, stream>>>(Wq, Wqp, HID);
  pack_b_kernel<<<(HID*HID + 255)/256, 256, 0, stream>>>(Wk, Wkp, HID);
  pack_b_kernel<<<(HID*HID + 255)/256, 256, 0, stream>>>(Wv, Wvp, HID);
  pack_b_kernel<<<(HID*HID + 255)/256, 256, 0, stream>>>(Wo, Wop, HID);

  k_embed_ln_gelu<<<ROWS/16, 256, 0, stream>>>(eidx, conf, etab, W1p, b1, lng, lnb, hb);
  k_attn_out<<<NB/4, 256, 0, stream>>>(hb, Wqp, Wkp, Wvp, Wop, eidx,
                                       lq1, lk1, lq2, lk2, subw, out);
}
